// HierarchicalGATEmbedding_37958920962378
// MI455X (gfx1250) — matrix operand order repaired and checked by reading, not compiled, not run on hardware
//
#include <hip/hip_runtime.h>

// Sizes from the reference
#define B_    8
#define S_    2048
#define H_    768
#define NH_   12
#define DH_   64
#define LEAF_ 8
#define NREL_ 100
#define TOK_  16            // tokens per block (one WMMA M-tile)
#define KSTEPS_ (H_/32)     // 24 k-steps of 32 for bf16 WMMA
#define NT_PER_WAVE_ 6      // 48 N-tiles / 8 waves

typedef __attribute__((ext_vector_type(16))) __bf16 v16bf;
typedef __attribute__((ext_vector_type(8)))  __bf16 v8bf;
typedef __attribute__((ext_vector_type(8)))  float  v8f;

static __device__ __forceinline__ __bf16 f2bf_rne(float x) {
  unsigned u = __builtin_bit_cast(unsigned, x);
  unsigned r = u + 0x7FFFu + ((u >> 16) & 1u);   // round-to-nearest-even
  return __builtin_bit_cast(__bf16, (unsigned short)(r >> 16));
}
static __device__ __forceinline__ float bf2f(__bf16 h) {
  unsigned u = ((unsigned)__builtin_bit_cast(unsigned short, h)) << 16;
  return __builtin_bit_cast(float, u);
}
static __device__ __forceinline__ void split_bf(float x, __bf16* hi, __bf16* lo) {
  __bf16 h = f2bf_rne(x);
  *hi = h;
  *lo = f2bf_rne(x - bf2f(h));
}

static __device__ __forceinline__ v8f wmma_bf(v16bf a, v16bf b, v8f c) {
  return __builtin_amdgcn_wmma_f32_16x16x32_bf16(
      /*neg_a=*/false, a, /*neg_b=*/false, b,
      /*c_mod=*/(short)0, c, /*reuse_a=*/false, /*reuse_b=*/false);
}

// A-fragment (16x32 bf16) load from row-major [16][768] LDS tile.
// lane: M = lane%16 ; element e -> K = kb + (lane/16)*16 + e
static __device__ __forceinline__ v16bf load_a_frag(const __bf16* p) {
  v8bf x = *(const v8bf*)(p);        // K .. K+7
  v8bf y = *(const v8bf*)(p + 8);    // K+8 .. K+15
  v16bf r;
#pragma unroll
  for (int i = 0; i < 8; ++i) { r[i] = x[i]; r[i + 8] = y[i]; }
  return r;
}

// ---------------------------------------------------------------------------
// Prep 1: transpose + hi/lo split Wq and Wo into bf16 tables.
// WT[n][k] layout so B-fragments (N = lane%16, K = kb + (lane/16)*16 + e)
// are 16 contiguous bf16 per lane.
// ---------------------------------------------------------------------------
__global__ __launch_bounds__(256) void prep_w(
    const float* __restrict__ Wq, const float* __restrict__ Wo,
    __bf16* __restrict__ qh, __bf16* __restrict__ ql,
    __bf16* __restrict__ oh, __bf16* __restrict__ ol) {
  int i = blockIdx.x * 256 + threadIdx.x;       // 0 .. 2*H*H-1
  const int NW = H_ * H_;
  int which = (i >= NW);
  int e = which ? (i - NW) : i;
  const float* W = which ? Wo : Wq;
  __bf16* ph = which ? oh : qh;
  __bf16* pl = which ? ol : ql;
  int k = e / H_, n = e - k * H_;
  __bf16 hi, lo;
  split_bf(W[e], &hi, &lo);
  ph[n * H_ + k] = hi;
  pl[n * H_ + k] = lo;
}

// ---------------------------------------------------------------------------
// Prep 2: Ktab/Vtab = rel_table @ Wk/Wv + bk/bv   (fp32, tiny GEMM)
// grid (NREL, 6), block 128 -> one (rel, 128-column slab) per block.
// ---------------------------------------------------------------------------
__global__ __launch_bounds__(128) void prep_kv(
    const float* __restrict__ relt,
    const float* __restrict__ Wk, const float* __restrict__ bk,
    const float* __restrict__ Wv, const float* __restrict__ bv,
    float* __restrict__ Kt, float* __restrict__ Vt) {
  __shared__ float row[H_];
  int r = blockIdx.x;
  for (int i = threadIdx.x; i < H_; i += 128) row[i] = relt[r * H_ + i];
  __syncthreads();
  int c = blockIdx.y * 128 + threadIdx.x;
  float dk = 0.f, dv = 0.f;
  for (int j = 0; j < H_; ++j) {
    float a = row[j];
    dk += a * Wk[j * H_ + c];
    dv += a * Wv[j * H_ + c];
  }
  Kt[r * H_ + c] = dk + bk[c];
  Vt[r * H_ + c] = dv + bv[c];
}

// ---------------------------------------------------------------------------
// GEMM tile: C[16x768] = A(16x768 bf16 hi/lo, LDS) x BT(768x768 bf16 hi/lo,
// global, pre-transposed) + bias (+ optional residual). 3-WMMA split product.
// ---------------------------------------------------------------------------
static __device__ __forceinline__ void gemm_tile(
    const __bf16* Ah, const __bf16* Al,
    const __bf16* __restrict__ BTh, const __bf16* __restrict__ BTl,
    const float* __restrict__ bias, float* Cout,
    int wave, int lane, const float* __restrict__ resid) {
  v8f acc[NT_PER_WAVE_];
#pragma unroll
  for (int j = 0; j < NT_PER_WAVE_; ++j)
#pragma unroll
    for (int r = 0; r < 8; ++r) acc[j][r] = 0.f;

  const int m    = lane & 15;
  const int kA   = (lane >> 4) * 16;   // A half-offset
  const int kB   = (lane >> 4) * 16;   // B half-offset
  for (int ks = 0; ks < KSTEPS_; ++ks) {
    const int kb = ks * 32;
    v16bf a_hi = load_a_frag(Ah + m * H_ + kb + kA);
    v16bf a_lo = load_a_frag(Al + m * H_ + kb + kA);
#pragma unroll
    for (int j = 0; j < NT_PER_WAVE_; ++j) {
      int n = (wave + 8 * j) * 16 + m;
      v16bf b_hi = *(const v16bf*)(BTh + (size_t)n * H_ + kb + kB);
      v16bf b_lo = *(const v16bf*)(BTl + (size_t)n * H_ + kb + kB);
      acc[j] = wmma_bf(a_hi, b_hi, acc[j]);
      acc[j] = wmma_bf(a_hi, b_lo, acc[j]);
      acc[j] = wmma_bf(a_lo, b_hi, acc[j]);
    }
  }
  const int mh = (lane >> 4) * 8;      // C/D: row r -> M = r + 8*(lane/16)
#pragma unroll
  for (int j = 0; j < NT_PER_WAVE_; ++j) {
    int n = (wave + 8 * j) * 16 + m;
    float bn = bias[n];
#pragma unroll
    for (int r = 0; r < 8; ++r) {
      int M = r + mh;
      float v = acc[j][r] + bn;
      if (resid) v += resid[M * H_ + n];
      Cout[M * H_ + n] = v;
    }
  }
}

// ---------------------------------------------------------------------------
// Fused: Q-proj (WMMA) -> masked 8-leaf attention (VALU) -> O-proj (WMMA)
// + residual -> LayerNorm.  One block = 16 tokens, 256 threads (8 waves).
// ---------------------------------------------------------------------------
__global__ __launch_bounds__(256) void fused(
    const float* __restrict__ text, const int* __restrict__ gs,
    const int* __restrict__ rel,
    const float* __restrict__ bq, const float* __restrict__ bo,
    const float* __restrict__ gamma, const float* __restrict__ beta,
    const __bf16* __restrict__ WqTh, const __bf16* __restrict__ WqTl,
    const __bf16* __restrict__ WoTh, const __bf16* __restrict__ WoTl,
    const float* __restrict__ Ktab, const float* __restrict__ Vtab,
    float* __restrict__ out) {
  __shared__ __align__(16) __bf16 Ahi[TOK_ * H_];   // text hi -> later ctx hi
  __shared__ __align__(16) __bf16 Alo[TOK_ * H_];   // text lo -> later ctx lo
  __shared__ __align__(16) float  Cf[TOK_ * H_];    // Q -> later out+residual
  __shared__ float mu_s[TOK_], rs_s[TOK_];

  const int tid  = threadIdx.x;
  const int wave = tid >> 5, lane = tid & 31;
  const int tok0 = blockIdx.x * TOK_;
  const float* xbase = text + (size_t)tok0 * H_;

  // Stage text tile as bf16 hi/lo
  for (int i = tid; i < TOK_ * H_; i += 256) {
    __bf16 hi, lo;
    split_bf(xbase[i], &hi, &lo);
    Ahi[i] = hi; Alo[i] = lo;
  }
  __syncthreads();

  // Q = X @ Wq + bq   -> Cf
  gemm_tile(Ahi, Alo, WqTh, WqTl, bq, Cf, wave, lane, nullptr);
  __syncthreads();

  // Attention: one thread per (token, head); 192 of 256 active
  if (tid < TOK_ * NH_) {
    const int token = tid / NH_;
    const int head  = tid - token * NH_;
    const int hoff  = head * DH_;
    const float* qp = Cf + token * H_ + hoff;
    const size_t g  = (size_t)(tok0 + token) * LEAF_;

    int   ridx[LEAF_];
    float sc[LEAF_];
#pragma unroll
    for (int l = 0; l < LEAF_; ++l) {
      int gv = gs[g + l];
      int r  = rel[g + l]; if (r < 0) r = 0;
      ridx[l] = r;
      const float4* kp = (const float4*)(Ktab + (size_t)r * H_ + hoff);
      const float4* qv = (const float4*)qp;
      float dot = 0.f;
#pragma unroll
      for (int d = 0; d < DH_ / 4; ++d) {
        float4 a = qv[d], b = kp[d];
        dot += a.x * b.x + a.y * b.y + a.z * b.z + a.w * b.w;
      }
      sc[l] = (gv != -1) ? dot * 0.125f : -10000.0f;   // 1/sqrt(64)
    }
    float mx = sc[0];
#pragma unroll
    for (int l = 1; l < LEAF_; ++l) mx = fmaxf(mx, sc[l]);
    float ssum = 0.f;
#pragma unroll
    for (int l = 0; l < LEAF_; ++l) { sc[l] = __expf(sc[l] - mx); ssum += sc[l]; }
    float inv = 1.0f / ssum;

    float cx[DH_];
#pragma unroll
    for (int d = 0; d < DH_; ++d) cx[d] = 0.f;
#pragma unroll
    for (int l = 0; l < LEAF_; ++l) {
      float p = sc[l] * inv;
      const float4* vp = (const float4*)(Vtab + (size_t)ridx[l] * H_ + hoff);
#pragma unroll
      for (int d = 0; d < DH_ / 4; ++d) {
        float4 v = vp[d];
        cx[4 * d + 0] += p * v.x; cx[4 * d + 1] += p * v.y;
        cx[4 * d + 2] += p * v.z; cx[4 * d + 3] += p * v.w;
      }
    }
    // ctx -> bf16 hi/lo (A-region is free after the Q-GEMM barrier)
#pragma unroll
    for (int d = 0; d < DH_; ++d) {
      __bf16 hi, lo;
      split_bf(cx[d], &hi, &lo);
      Ahi[token * H_ + hoff + d] = hi;
      Alo[token * H_ + hoff + d] = lo;
    }
  }
  __syncthreads();

  // out = ctx @ Wo + bo + residual(text)  -> Cf  (overwrites Q)
  gemm_tile(Ahi, Alo, WoTh, WoTl, bo, Cf, wave, lane, xbase);
  __syncthreads();

  // LayerNorm statistics (one thread per token)
  if (tid < TOK_) {
    const float* row = Cf + tid * H_;
    float s = 0.f;
    for (int i = 0; i < H_; ++i) s += row[i];
    float mu = s * (1.0f / H_);
    float v = 0.f;
    for (int i = 0; i < H_; ++i) { float d = row[i] - mu; v += d * d; }
    v *= (1.0f / H_);
    mu_s[tid] = mu;
    rs_s[tid] = rsqrtf(v + 1e-5f);
  }
  __syncthreads();

  float* obase = out + (size_t)tok0 * H_;
  for (int i = tid; i < TOK_ * H_; i += 256) {
    int t = i / H_, c = i - t * H_;
    obase[i] = (Cf[i] - mu_s[t]) * rs_s[t] * gamma[c] + beta[c];
  }
}

// ---------------------------------------------------------------------------
extern "C" void kernel_launch(void* const* d_in, const int* in_sizes, int n_in,
                              void* d_out, int out_size, void* d_ws, size_t ws_size,
                              hipStream_t stream) {
  (void)in_sizes; (void)n_in; (void)out_size; (void)ws_size;
  const float* text  = (const float*)d_in[0];
  const int*   gsi   = (const int*)d_in[1];
  const int*   reli  = (const int*)d_in[2];
  const float* reltb = (const float*)d_in[3];
  const float* Wq    = (const float*)d_in[4];
  const float* bq    = (const float*)d_in[5];
  const float* Wk    = (const float*)d_in[6];
  const float* bk    = (const float*)d_in[7];
  const float* Wv    = (const float*)d_in[8];
  const float* bv    = (const float*)d_in[9];
  const float* Wo    = (const float*)d_in[10];
  const float* bo    = (const float*)d_in[11];
  const float* gam   = (const float*)d_in[12];
  const float* bet   = (const float*)d_in[13];
  float* outp = (float*)d_out;

  // Workspace layout (bytes)
  char* ws = (char*)d_ws;
  const size_t WSZ = (size_t)H_ * H_ * sizeof(unsigned short);  // 1,179,648
  __bf16* WqTh = (__bf16*)(ws + 0 * WSZ);
  __bf16* WqTl = (__bf16*)(ws + 1 * WSZ);
  __bf16* WoTh = (__bf16*)(ws + 2 * WSZ);
  __bf16* WoTl = (__bf16*)(ws + 3 * WSZ);
  float*  Ktab = (float*)(ws + 4 * WSZ);
  float*  Vtab = (float*)(ws + 4 * WSZ + (size_t)NREL_ * H_ * sizeof(float));

  prep_w<<<(2 * H_ * H_) / 256, 256, 0, stream>>>(Wq, Wo, WqTh, WqTl, WoTh, WoTl);
  prep_kv<<<dim3(NREL_, H_ / 128), 128, 0, stream>>>(reltb, Wk, bk, Wv, bv, Ktab, Vtab);
  fused<<<(B_ * S_) / TOK_, 256, 0, stream>>>(
      text, gsi, reli, bq, bo, gam, bet,
      WqTh, WqTl, WoTh, WoTl, Ktab, Vtab, outp);
}
